// YOLOObjectDetector_14542759264831
// MI455X (gfx1250) — matrix operand order repaired and checked by reading, not compiled, not run on hardware
//
#include <hip/hip_runtime.h>
#include <hip/hip_bf16.h>

// ---------------------------------------------------------------------------
// YOLO head for MI455X (gfx1250, wave32, WMMA).
//   pack:  x(256,1280,14,14)[::2,::2] -> Xp[12544,1280] bf16   (im2row for 1x1 conv)
//   gemm0: Xp x conv_w^T  -> H1 bf16, scattered to (b, o*49+s) layout
//   gemm1: H1 x fc1_w^T   -> H2 bf16, leaky-relu
//   gemm2: H2 x fc2_w^T   -> H3 f32,  sigmoid
//   nms:   per-image decode of 98 boxes + NMS + top-10
// All matmuls: bf16 A/B (converted in-flight from f32 weights), f32 accum via
// v_wmma_f32_16x16x32_bf16. Memory-bound on f32 weight streams (~128MB),
// so weights are read exactly once.
// ---------------------------------------------------------------------------

typedef __attribute__((ext_vector_type(16))) __bf16 v16bf;
typedef __attribute__((ext_vector_type(8)))  __bf16 v8bf;
typedef __attribute__((ext_vector_type(8)))  float  v8f;

__device__ __forceinline__ __bf16 f2bf(float f) {
    unsigned u = __builtin_bit_cast(unsigned, f);
    u += 0x7FFFu + ((u >> 16) & 1u);           // round-to-nearest-even
    return __builtin_bit_cast(__bf16, (unsigned short)(u >> 16));
}

// ------------------------------ pack kernel --------------------------------
// Xp[m=b*49+h*7+w][c] = x[b][c][2h][2w], converted to bf16. Coalesced writes.
__global__ __launch_bounds__(256) void pack_x_kernel(
    const float* __restrict__ x, __bf16* __restrict__ Xp)
{
    size_t idx = (size_t)blockIdx.x * 256 + threadIdx.x;   // over 12544*1280
    int c = (int)(idx % 1280);
    int m = (int)(idx / 1280);
    int b = m / 49, s = m - b * 49;
    int hh = s / 7, ww = s - hh * 7;
    size_t src = (((size_t)b * 1280 + c) * 14 + 2 * hh) * 14 + 2 * ww;
    Xp[idx] = f2bf(x[src]);
}

// ------------------------------ GEMM kernel --------------------------------
// C[M,Nval] = A[M,K](bf16) * W[Nval,K](f32)^T + bias, tile 64x128x32.
// MODE 0: conv scatter -> bf16 out[b*6272 + n*49 + s],  m = b*49+s
// MODE 1: leaky-relu   -> bf16 out[m*Nval + n]
// MODE 2: sigmoid      -> f32  out[m*Nval + n]
template <int MODE>
__global__ __launch_bounds__(256) void gemm_wmma_bf16(
    const __bf16* __restrict__ A, const float* __restrict__ W,
    const float* __restrict__ bias, void* __restrict__ outp,
    int M, int Nval, int K)
{
    constexpr int BM = 64, BN = 128, BK = 32, LDA = 40, LDB = 40;
    __shared__ __align__(16) __bf16 As[BM * LDA];
    __shared__ __align__(16) __bf16 Bs[BN * LDB];

    const int tid  = threadIdx.x;
    const int lane = tid & 31;
    const int wid  = tid >> 5;                 // 8 waves
    const int wm   = (wid >> 2) * 32;          // 2x4 wave grid, 32x32 each
    const int wn   = (wid & 3) * 32;
    const int m0   = blockIdx.y * BM;
    const int n0   = blockIdx.x * BN;

    v8f acc[2][2] = {};

    const int aRow = tid >> 2;                 // A tile: 64x32 bf16, 16B/thread
    const int aCol = (tid & 3) * 8;
    const int bRow = tid >> 1;                 // W tile: 128x32 f32, 16 f32/thread
    const int bCol = (tid & 1) * 16;
    const int mrow = lane & 15;
    const int hi   = lane >> 4;

    for (int k0 = 0; k0 < K; k0 += BK) {
        // cooperative A-tile load (already bf16)
        *reinterpret_cast<uint4*>(&As[aRow * LDA + aCol]) =
            *reinterpret_cast<const uint4*>(A + (size_t)(m0 + aRow) * K + k0 + aCol);
        // cooperative W-tile load + f32->bf16 convert
        {
            int gn = n0 + bRow;
            union { __bf16 b[16]; uint4 u[2]; } cv;
            if (gn < Nval) {
                const float4* src =
                    reinterpret_cast<const float4*>(W + (size_t)gn * K + k0 + bCol);
#pragma unroll
                for (int q = 0; q < 4; ++q) {
                    float4 f = src[q];
                    cv.b[q * 4 + 0] = f2bf(f.x); cv.b[q * 4 + 1] = f2bf(f.y);
                    cv.b[q * 4 + 2] = f2bf(f.z); cv.b[q * 4 + 3] = f2bf(f.w);
                }
                if (k0 + BK < K)  // pull next tile toward L2 (global_prefetch_b8)
                    __builtin_prefetch(W + (size_t)gn * K + k0 + BK + bCol, 0, 1);
            } else {
#pragma unroll
                for (int q = 0; q < 16; ++q)
                    cv.b[q] = __builtin_bit_cast(__bf16, (unsigned short)0);
            }
            uint4* dst = reinterpret_cast<uint4*>(&Bs[bRow * LDB + bCol]);
            dst[0] = cv.u[0]; dst[1] = cv.u[1];
        }
        __syncthreads();

        // Fragment loads per ISA wave32 layouts:
        //  A 16x32: lane<16 -> K 0..7 / 16..23 ; lane>=16 -> K 8..15 / 24..31
        //  B 32x16: lane&15 = N, lane<16 -> K 0..7 / 16..23 ; lane>=16 -> K 8..15 / 24..31
        v16bf af[2], bfg[2];
#pragma unroll
        for (int i = 0; i < 2; ++i) {
            const __bf16* ar = &As[(wm + i * 16 + mrow) * LDA];
            union { v16bf v; v8bf h[2]; } ua;
            ua.h[0] = *reinterpret_cast<const v8bf*>(ar + hi * 8);
            ua.h[1] = *reinterpret_cast<const v8bf*>(ar + 16 + hi * 8);
            af[i] = ua.v;
            const __bf16* br = &Bs[(wn + i * 16 + mrow) * LDB];
            union { v16bf v; v8bf h[2]; } ub;
            ub.h[0] = *reinterpret_cast<const v8bf*>(br + hi * 8);
            ub.h[1] = *reinterpret_cast<const v8bf*>(br + 16 + hi * 8);
            bfg[i] = ub.v;
        }
#pragma unroll
        for (int i = 0; i < 2; ++i)
#pragma unroll
            for (int j = 0; j < 2; ++j)
                acc[i][j] = __builtin_amdgcn_wmma_f32_16x16x32_bf16(
                    false, af[i], false, bfg[j], (short)0, acc[i][j], false, false);
        __syncthreads();
    }

    // Epilogue: C 16x16 f32 layout -> VGPR r: lanes0-15 M=r, lanes16-31 M=r+8
    const int crow = hi * 8;
    const int ccol = mrow;
#pragma unroll
    for (int i = 0; i < 2; ++i) {
#pragma unroll
        for (int j = 0; j < 2; ++j) {
            int n = n0 + wn + j * 16 + ccol;
            float bv = (n < Nval) ? bias[n] : 0.f;
#pragma unroll
            for (int r = 0; r < 8; ++r) {
                int m = m0 + wm + i * 16 + crow + r;
                if (m < M && n < Nval) {
                    float v = acc[i][j][r] + bv;
                    if (MODE == 1) v = (v >= 0.f) ? v : 0.1f * v;
                    if (MODE == 2) {
                        v = 1.0f / (1.0f + __expf(-v));
                        reinterpret_cast<float*>(outp)[(size_t)m * Nval + n] = v;
                    } else if (MODE == 0) {
                        int b = m / 49, s = m - b * 49;
                        reinterpret_cast<__bf16*>(outp)[(size_t)b * 6272 + n * 49 + s] = f2bf(v);
                    } else {
                        reinterpret_cast<__bf16*>(outp)[(size_t)m * Nval + n] = f2bf(v);
                    }
                }
            }
        }
    }
}

// --------------------------- decode + NMS kernel ---------------------------
// One block per image (256 blocks, 128 threads). 98 boxes, O(98^2) in LDS.
__global__ __launch_bounds__(128) void decode_nms_kernel(
    const float* __restrict__ H3, float* __restrict__ out)
{
    const int b = blockIdx.x, t = threadIdx.x;
    __shared__ float sh[1470];
    __shared__ float bx1[98], by1[98], bx2[98], by2[98], bcf[98];
    __shared__ int   blab[98], sord[98], skeep[98];
    __shared__ float sx1[98], sy1[98], sx2[98], sy2[98], scf[98];
    __shared__ int   slb[98];

    const float* hsrc = H3 + (size_t)b * 1470;
    for (int i = t; i < 1470; i += 128) sh[i] = hsrc[i];
    __syncthreads();

    if (t < 98) {
        int cell = t >> 1;
        int gi = cell / 7, gj = cell - gi * 7;
        float cx = sh[t * 4 + 0] * 64.f + gi * 64.f;
        float cy = sh[t * 4 + 1] * 64.f + gj * 64.f;
        float w  = sh[t * 4 + 2] * 448.f;
        float hh = sh[t * 4 + 3] * 448.f;
        bx1[t] = fminf(fmaxf(cx - 0.5f * w, 0.f), 448.f);
        by1[t] = fminf(fmaxf(cy - 0.5f * hh, 0.f), 448.f);
        bx2[t] = fminf(fmaxf(cx + 0.5f * w, 0.f), 448.f);
        by2[t] = fminf(fmaxf(cy + 0.5f * hh, 0.f), 448.f);
        bcf[t] = sh[392 + t];
        int base = 490 + cell * 20, best = 0;
        float bv = sh[base];
        for (int c = 1; c < 20; ++c) {
            float v = sh[base + c];
            if (v > bv) { bv = v; best = c; }
        }
        blab[t] = best;
    }
    __syncthreads();

    // stable argsort(-conf) via rank counting
    if (t < 98) {
        float cv = bcf[t];
        int r = 0;
        for (int m = 0; m < 98; ++m) {
            float cm = bcf[m];
            if (cm > cv || (cm == cv && m < t)) ++r;
        }
        sord[r] = t;
    }
    __syncthreads();
    if (t < 98) {
        int s = sord[t];
        sx1[t] = bx1[s]; sy1[t] = by1[s]; sx2[t] = bx2[s]; sy2[t] = by2[s];
        scf[t] = bcf[s]; slb[t] = blab[s];
        skeep[t] = (scf[t] > 0.1f) ? 1 : 0;
    }
    __syncthreads();

    // sequential greedy suppression, barrier-stepped
    for (int i = 0; i < 98; ++i) {
        if (t < 98 && t > i && skeep[i] && skeep[t]) {
            float ai = fmaxf(sx2[i] - sx1[i], 0.f) * fmaxf(sy2[i] - sy1[i], 0.f);
            float at = fmaxf(sx2[t] - sx1[t], 0.f) * fmaxf(sy2[t] - sy1[t], 0.f);
            float ix1 = fmaxf(sx1[i], sx1[t]), iy1 = fmaxf(sy1[i], sy1[t]);
            float ix2 = fminf(sx2[i], sx2[t]), iy2 = fminf(sy2[i], sy2[t]);
            float inter = fmaxf(ix2 - ix1, 0.f) * fmaxf(iy2 - iy1, 0.f);
            float un = ai + at - inter;
            float iou = (un > 0.f) ? inter / un : 0.f;
            if (iou > 0.7f) skeep[t] = 0;
        }
        __syncthreads();
    }

    // top-10 kept (already conf-sorted); zeros elsewhere
    if (t == 0) {
        float* ob = out + (size_t)b * 40;
        float* ol = out + 10240 + (size_t)b * 10;
        float* oc = out + 12800 + (size_t)b * 10;
        int cnt = 0;
        for (int r = 0; r < 98 && cnt < 10; ++r) {
            if (skeep[r]) {
                ob[cnt * 4 + 0] = sx1[r]; ob[cnt * 4 + 1] = sy1[r];
                ob[cnt * 4 + 2] = sx2[r]; ob[cnt * 4 + 3] = sy2[r];
                ol[cnt] = (float)slb[r]; oc[cnt] = scf[r];
                ++cnt;
            }
        }
        for (; cnt < 10; ++cnt) {
            ob[cnt * 4 + 0] = 0.f; ob[cnt * 4 + 1] = 0.f;
            ob[cnt * 4 + 2] = 0.f; ob[cnt * 4 + 3] = 0.f;
            ol[cnt] = 0.f; oc[cnt] = 0.f;
        }
    }
}

// ------------------------------ launcher -----------------------------------
extern "C" void kernel_launch(void* const* d_in, const int* in_sizes, int n_in,
                              void* d_out, int out_size, void* d_ws, size_t ws_size,
                              hipStream_t stream)
{
    (void)in_sizes; (void)n_in; (void)out_size; (void)ws_size;
    const float* x      = (const float*)d_in[0];
    const float* conv_w = (const float*)d_in[1];
    const float* conv_b = (const float*)d_in[2];
    const float* fc1_w  = (const float*)d_in[3];
    const float* fc1_b  = (const float*)d_in[4];
    const float* fc2_w  = (const float*)d_in[5];
    const float* fc2_b  = (const float*)d_in[6];

    char* ws = (char*)d_ws;
    __bf16* Xp = (__bf16*)(ws);                                   // 12544*1280*2 = 32,112,640
    __bf16* H1 = (__bf16*)(ws + 32112640);                        // 256*6272*2  =  3,211,264
    __bf16* H2 = (__bf16*)(ws + 32112640 + 3211264);              // 256*4096*2  =  2,097,152
    float*  H3 = (float*)(ws + 32112640 + 3211264 + 2097152);     // 256*1470*4  =  1,505,280

    pack_x_kernel<<<62720, 256, 0, stream>>>(x, Xp);
    // conv1x1: [12544,1280] x [1280,128] -> scatter to fc1 layout
    gemm_wmma_bf16<0><<<dim3(1, 196),  256, 0, stream>>>(Xp, conv_w, conv_b, (void*)H1, 12544, 128, 1280);
    // fc1: [256,6272] x [6272,4096], leaky-relu
    gemm_wmma_bf16<1><<<dim3(32, 4),   256, 0, stream>>>(H1, fc1_w, fc1_b, (void*)H2, 256, 4096, 6272);
    // fc2: [256,4096] x [4096,1470], sigmoid
    gemm_wmma_bf16<2><<<dim3(12, 4),   256, 0, stream>>>(H2, fc2_w, fc2_b, (void*)H3, 256, 1470, 4096);
    decode_nms_kernel<<<256, 128, 0, stream>>>(H3, (float*)d_out);
}
